// GraphSwinPrePartD6BD8A_1073741824682
// MI455X (gfx1250) — matrix operand order repaired and checked by reading, not compiled, not run on hardware
//
#include <hip/hip_runtime.h>
#include <hip/hip_bf16.h>
#include <math.h>

// ---------------------------------------------------------------------------
// GraphSwinPrePart for MI455X (gfx1250, wave32, WMMA)
//
// Key algebraic fact: reg-context and shifted-context are constant over 3x3
// pixel blocks, so fuse/out_proj GEMMs only need to run on 32*32*32 = 32768
// block tokens instead of 294912 pixels.  The kernel is HBM-bound on the
// single read of h_pixel (151MB) and single write of out (151MB) ~ 13us at
// 23.3 TB/s; all GEMM work (~2.7 GFLOP) goes through v_wmma_f32_16x16x32_bf16.
// ---------------------------------------------------------------------------

typedef __attribute__((ext_vector_type(16))) __bf16 v16bf;
typedef __attribute__((ext_vector_type(8)))  float  v8f;

union ABfrag { unsigned int u[8]; v16bf h; };

#define HID   128
#define HH    96
#define WW    96
#define BATCH 32

// ---------------------------------------------------------------------------
// Kernel 1: 3x3-block partial sums.  psum[b][bi][bj][d], bi,bj in [0,32)
// threads: 32*1024*128 = 4194304, fully coalesced over d.
// ---------------------------------------------------------------------------
__global__ __launch_bounds__(256) void k_block_sum(const float* __restrict__ in,
                                                   float* __restrict__ psum) {
  size_t t = (size_t)blockIdx.x * 256 + threadIdx.x;      // < 4194304
  int d   = (int)(t & 127);
  int blk = (int)(t >> 7);                                 // 0..32767
  int b   = blk >> 10;
  int r   = blk & 1023;
  int bi  = r >> 5, bj = r & 31;
  const float* base = in + (((size_t)b * (HH * WW)) + (size_t)(bi * 3) * WW + bj * 3) * HID + d;
  float s = 0.f;
#pragma unroll
  for (int dr = 0; dr < 3; ++dr)
#pragma unroll
    for (int dc = 0; dc < 3; ++dc)
      s += base[(size_t)(dr * WW + dc) * HID];
  psum[t] = s;
}

// ---------------------------------------------------------------------------
// Kernel 2: regular + shifted window means from block partial sums.
// tokmean[which][b][w][d], which=0 regular, which=1 shifted. 2097152 threads.
// Regular window (wi,wj) = blocks (2wi..2wi+1, 2wj..2wj+1).
// Shifted window (wi,wj) = blocks (2wi+1, 2wi+2 mod 32) x (2wj+1, 2wj+2 mod 32).
// ---------------------------------------------------------------------------
__global__ __launch_bounds__(256) void k_win_mean(const float* __restrict__ psum,
                                                  float* __restrict__ tokmean) {
  size_t t = (size_t)blockIdx.x * 256 + threadIdx.x;       // < 2097152
  int d  = (int)(t & 127);
  int r  = (int)(t >> 7);                                   // 0..16383
  int which = r >> 13;
  int rr = r & 8191;
  int b  = rr >> 8;
  int w  = rr & 255;
  int wi = w >> 4, wj = w & 15;
  int i0, i1, j0, j1;
  if (which == 0) { i0 = 2 * wi;            i1 = 2 * wi + 1;
                    j0 = 2 * wj;            j1 = 2 * wj + 1; }
  else            { i0 = (2 * wi + 1) & 31; i1 = (2 * wi + 2) & 31;
                    j0 = (2 * wj + 1) & 31; j1 = (2 * wj + 2) & 31; }
  const float* P = psum + ((size_t)b << 17) + d;            // b*1024*128
  float s = P[(size_t)(i0 * 32 + j0) * HID] + P[(size_t)(i0 * 32 + j1) * HID] +
            P[(size_t)(i1 * 32 + j0) * HID] + P[(size_t)(i1 * 32 + j1) * HID];
  tokmean[t] = s * (1.f / 36.f);
}

// ---------------------------------------------------------------------------
// Kernel 3: fused  Y = GELU(LayerNorm(X @ W + bias))  for X[M,128], W[128,128].
// One wave -> one 16-row M tile x all 128 N columns (8 f32 accumulators).
// 8 waves / block -> 128 rows / block.  K=128 = 4 steps of 16x16x32 bf16 WMMA.
// W is staged in LDS transposed & bf16-pair-packed with pad so every B-frag
// is two aligned ds_load_b128.  All 4 A-fragments are built up front (global
// loads issue in clauses and conversion overlaps the WMMAs); B-fragment LDS
// loads are software-pipelined one tile ahead so waits are dscnt<=2 instead
// of dscnt==0.  LayerNorm reduction done in the C-layout via shfl_xor over
// the 16-lane half groups.
// ---------------------------------------------------------------------------
__global__ __launch_bounds__(256) void k_gemm_ln_gelu(const float* __restrict__ X,
                                                      const float* __restrict__ Wg,
                                                      const float* __restrict__ bias,
                                                      const float* __restrict__ gamma,
                                                      const float* __restrict__ beta,
                                                      float* __restrict__ Y, int M) {
  __shared__ unsigned int sW[128 * 68];   // [n][k2] bf16 pair (k=2*k2, 2*k2+1), pad 68
  const int t = threadIdx.x;

  // Stage W: consecutive threads read consecutive n (coalesced in n).
  for (int i = t; i < 8192; i += 256) {
    int k2 = i >> 7, n = i & 127;
    unsigned lo = __float_as_uint(Wg[(size_t)(2 * k2) * HID + n]) >> 16;
    unsigned hi = __float_as_uint(Wg[(size_t)(2 * k2 + 1) * HID + n]) & 0xffff0000u;
    sW[n * 68 + k2] = lo | hi;
  }
  __syncthreads();

  const int wave = t >> 5, lane = t & 31;
  const int half = lane >> 4, l15 = lane & 15;
  const int rowBase = blockIdx.x * 128 + wave * 16;
  if (rowBase >= M) return;
  const float* xrow = X + (size_t)(rowBase + l15) * HID;
  const unsigned int* sWl = &sW[l15 * 68 + half * 4];

  // Build all 4 A fragments up front: lane (l15,half) holds row l15; K layout
  // per ISA 16-bit A 16x32 table (k = kt*32 + (p>>2)*16 + half*8 + (p&3)*2).
  ABfrag a[4];
#pragma unroll
  for (int kt = 0; kt < 4; ++kt) {
#pragma unroll
    for (int p = 0; p < 8; ++p) {
      int k = kt * 32 + ((p >> 2) << 4) + (half << 3) + ((p & 3) << 1);
      float2 xv = *(const float2*)(xrow + k);
      a[kt].u[p] = (__float_as_uint(xv.x) >> 16) | (__float_as_uint(xv.y) & 0xffff0000u);
    }
  }

  v8f acc[8];
#pragma unroll
  for (int nt = 0; nt < 8; ++nt)
#pragma unroll
    for (int v = 0; v < 8; ++v) acc[nt][v] = 0.f;

#pragma unroll
  for (int kt = 0; kt < 4; ++kt) {
    // B-fragment pipeline: load tile nt+1 while WMMA consumes tile nt.
    ABfrag bcur, bnxt;
    {
      const unsigned int* bp = sWl + kt * 16;               // nt = 0
      uint4 b0 = *(const uint4*)bp;
      uint4 b1 = *(const uint4*)(bp + 8);
      bcur.u[0] = b0.x; bcur.u[1] = b0.y; bcur.u[2] = b0.z; bcur.u[3] = b0.w;
      bcur.u[4] = b1.x; bcur.u[5] = b1.y; bcur.u[6] = b1.z; bcur.u[7] = b1.w;
    }
#pragma unroll
    for (int nt = 0; nt < 8; ++nt) {
      if (nt < 7) {
        const unsigned int* bp = sWl + (nt + 1) * 16 * 68 + kt * 16;
        uint4 b0 = *(const uint4*)bp;
        uint4 b1 = *(const uint4*)(bp + 8);
        bnxt.u[0] = b0.x; bnxt.u[1] = b0.y; bnxt.u[2] = b0.z; bnxt.u[3] = b0.w;
        bnxt.u[4] = b1.x; bnxt.u[5] = b1.y; bnxt.u[6] = b1.z; bnxt.u[7] = b1.w;
      }
      acc[nt] = __builtin_amdgcn_wmma_f32_16x16x32_bf16(
          false, a[kt].h, false, bcur.h, (short)0, acc[nt], false, false);
      bcur = bnxt;
    }
  }

  // Epilogue: +bias, LayerNorm over 128 dims, exact GELU, store.
  float bia[8], gam[8], bet[8];
#pragma unroll
  for (int nt = 0; nt < 8; ++nt) {
    int n = nt * 16 + l15;
    bia[nt] = bias[n]; gam[nt] = gamma[n]; bet[nt] = beta[n];
  }
#pragma unroll
  for (int nt = 0; nt < 8; ++nt)
#pragma unroll
    for (int v = 0; v < 8; ++v) acc[nt][v] += bia[nt];

#pragma unroll
  for (int v = 0; v < 8; ++v) {
    float s = 0.f, s2 = 0.f;
#pragma unroll
    for (int nt = 0; nt < 8; ++nt) { float x = acc[nt][v]; s += x; s2 += x * x; }
    // reduce across the 16 lanes of this half group (rows identical there)
#pragma unroll
    for (int m = 1; m <= 8; m <<= 1) {
      s  += __shfl_xor(s,  m, 32);
      s2 += __shfl_xor(s2, m, 32);
    }
    float mean = s * (1.f / 128.f);
    float var  = s2 * (1.f / 128.f) - mean * mean;
    float rs   = rsqrtf(var + 1e-5f);
    int row = rowBase + v + 8 * half;
    float* yrow = Y + (size_t)row * HID;
#pragma unroll
    for (int nt = 0; nt < 8; ++nt) {
      float y = (acc[nt][v] - mean) * rs * gam[nt] + bet[nt];
      y = 0.5f * y * (1.f + erff(y * 0.70710678118654752f));   // exact GELU
      yrow[nt * 16 + l15] = y;
    }
  }
}

// ---------------------------------------------------------------------------
// Kernel 4: ctx_in[b][bi][bj] = tok_reg[b, bi/2, bj/2] + tok_sh[b, ((bi-1)%32)/2, ((bj-1)%32)/2]
// (roll(+3) means pixel block bi sees shifted window ((bi-1) mod 32)/2)
// ---------------------------------------------------------------------------
__global__ __launch_bounds__(256) void k_ctx_combine(const float* __restrict__ tok,
                                                     float* __restrict__ ctxin) {
  size_t t = (size_t)blockIdx.x * 256 + threadIdx.x;       // < 4194304
  int d   = (int)(t & 127);
  int r   = (int)(t >> 7);                                  // 0..32767
  int b   = r >> 10;
  int blk = r & 1023;
  int bi  = blk >> 5, bj = blk & 31;
  int wreg = (bi >> 1) * 16 + (bj >> 1);
  int wsh  = (((bi + 31) & 31) >> 1) * 16 + (((bj + 31) & 31) >> 1);
  float a = tok[(((size_t)b * 256 + wreg) << 7) + d];           // which=0
  float c = tok[(((size_t)(32 + b) * 256 + wsh) << 7) + d];     // which=1
  ctxin[t] = a + c;
}

// ---------------------------------------------------------------------------
// Kernel 5: broadcast each block token to its 3x3 pixels, apply node mask.
// Fully coalesced 151MB store.
// ---------------------------------------------------------------------------
__global__ __launch_bounds__(256) void k_broadcast_mask(const float* __restrict__ ctx2,
                                                        const float* __restrict__ mask,
                                                        float* __restrict__ out) {
  size_t t = (size_t)blockIdx.x * 256 + threadIdx.x;       // < 37748736
  int d    = (int)(t & 127);
  size_t p = t >> 7;                                        // pixel 0..294911
  int b  = (int)(p / (HH * WW));
  int rc = (int)(p % (HH * WW));
  int rr = rc / WW, cc = rc % WW;
  size_t blk = (size_t)b * 1024 + (size_t)(rr / 3) * 32 + (cc / 3);
  out[t] = ctx2[(blk << 7) + d] * mask[p];
}

// ---------------------------------------------------------------------------
// Launch.  Workspace layout (floats), with reuse:
//   [0        .. 4194304)  psum      -> later ctx1
//   [4194304  .. 6291456)  tokmean   -\
//   [6291456  .. 8388608)  tokproj   -/-> later ctx2 (4194304 span)
//   [8388608  .. 12582912) ctx_in
// Total: 48 MB.
// ---------------------------------------------------------------------------
extern "C" void kernel_launch(void* const* d_in, const int* in_sizes, int n_in,
                              void* d_out, int out_size, void* d_ws, size_t ws_size,
                              hipStream_t stream) {
  const float* h_pixel = (const float*)d_in[0];
  const float* mask    = (const float*)d_in[1];
  const float* Wt  = (const float*)d_in[2];
  const float* bt  = (const float*)d_in[3];
  const float* gt  = (const float*)d_in[4];
  const float* bgt = (const float*)d_in[5];
  const float* Wf  = (const float*)d_in[6];
  const float* bf  = (const float*)d_in[7];
  const float* gf  = (const float*)d_in[8];
  const float* bgf = (const float*)d_in[9];
  const float* Wo  = (const float*)d_in[10];
  const float* bo  = (const float*)d_in[11];
  const float* go  = (const float*)d_in[12];
  const float* bgo = (const float*)d_in[13];
  float* out = (float*)d_out;

  float* ws      = (float*)d_ws;
  float* psum    = ws;                 // 4194304 f
  float* tokmean = ws + 4194304;       // 2097152 f
  float* tokp    = ws + 6291456;       // 2097152 f
  float* ctxin   = ws + 8388608;       // 4194304 f
  float* ctx1    = psum;               // reuse (psum dead)
  float* ctx2    = tokmean;            // reuse (tokmean+tokp dead), 4194304 f span

  k_block_sum<<<16384, 256, 0, stream>>>(h_pixel, psum);
  k_win_mean<<<8192, 256, 0, stream>>>(psum, tokmean);
  k_gemm_ln_gelu<<<128, 256, 0, stream>>>(tokmean, Wt, bt, gt, bgt, tokp, 16384);
  k_ctx_combine<<<16384, 256, 0, stream>>>(tokp, ctxin);
  k_gemm_ln_gelu<<<256, 256, 0, stream>>>(ctxin, Wf, bf, gf, bgf, ctx1, 32768);
  k_gemm_ln_gelu<<<256, 256, 0, stream>>>(ctx1, Wo, bo, go, bgo, ctx2, 32768);
  k_broadcast_mask<<<147456, 256, 0, stream>>>(ctx2, mask, out);
}
